// EMAQuantizer_83846351553221
// MI455X (gfx1250) — matrix operand order repaired and checked by reading, not compiled, not run on hardware
//
#include <hip/hip_runtime.h>

typedef __attribute__((ext_vector_type(16))) __bf16 v16bf;
typedef __attribute__((ext_vector_type(8)))  float  v8f;
typedef __attribute__((ext_vector_type(4)))  unsigned v4u;
typedef __attribute__((ext_vector_type(8)))  unsigned v8u;

#define NE      1024
#define CDIM    64
#define NPIX    65536      // 16*64*64 pixels
#define ROWS    128        // pixel rows per workgroup
#define ECHUNK  64         // codes staged per chunk (4 tiles of 16)
#define ZS_STRIDE 136      // 128 + 8 pad (== TDM pad_interval 128 DW / pad_amount 8 DW)

// d_out flat layout (reference tuple order)
#define OFF_ZQ   0
#define OFF_LOSS 4194304
#define OFF_IDX  4194305
#define OFF_NEMB 4210689
#define OFF_NCS  4276225
#define OFF_NEA  4277249

// workspace layout (floats)
#define WS_CNT   0
#define WS_ESUM  1024
#define WS_N     66560
#define WS_SSE   66561
#define WS_ZERON 66562

__device__ __forceinline__ v8f wmma_bf16(v16bf a, v16bf b, v8f c) {
    return __builtin_amdgcn_wmma_f32_16x16x32_bf16(false, a, false, b, (short)0, c, false, false);
}

__global__ __launch_bounds__(256) void vq_wmma_argmin_kernel(
    const float* __restrict__ z, const float* __restrict__ emb,
    float* __restrict__ out, float* __restrict__ ws)
{
    __shared__ float zs[CDIM * ZS_STRIDE];   // zs[c][r]
    __shared__ v16bf bfrag[256];             // 8KB: [tile][kc][hi/lo][lane] pre-packed B fragments
    __shared__ float esq_s[NE];              // |e|^2 for all codes
    __shared__ int   best_s[ROWS];
    __shared__ float wred[8];

    const int tid = threadIdx.x;
    const int n0  = blockIdx.x * ROWS;       // first flat pixel of this tile
    const int b   = n0 >> 12;                // n = b*4096 + h*64 + w
    const int o0  = n0 & 4095;               // rows are globally contiguous per channel
    const float* zb = z + b * 262144 + o0;   // + c*4096 + r

    const unsigned lds_base = (unsigned)(unsigned long long)(&zs[0]);

    // ---- stage z tile channels 0..31 with one TDM descriptor (TENSORcnt path) ----
    // 2D tile: tile_dim0=128 (contiguous), tile_dim1=32 rows, row stride 4096 elems,
    // LDS pad 8 DWORDs after every 128 DWORDs -> zs row stride 136.
    if (tid < 32) {
        unsigned long long ga = (unsigned long long)zb;
        v4u g0 = { 1u,                                   // count=1 (valid descriptor)
                   lds_base,                             // lds_addr
                   (unsigned)ga,                         // global_addr[31:0]
                   ((unsigned)(ga >> 32) & 0x1FFFFFFu) | 0x80000000u };  // addr[56:32] | type=2
        v8u g1 = { (2u << 16) | (1u << 20) | (6u << 22) | (7u << 25),  // 4B elems, pad en, 128DW/8DW
                   4096u << 16,                          // tensor_dim0 = 4096 (lo16 in [31:16])
                   64u << 16,                            // tensor_dim0 hi=0 | tensor_dim1 = 64
                   128u << 16,                           // tensor_dim1 hi=0 | tile_dim0 = 128
                   32u,                                  // tile_dim1 = 32, tile_dim2 = 0
                   4096u,                                // tensor_dim0_stride lo32
                   0u, 0u };
        asm volatile("tensor_load_to_lds %0, %1" :: "s"(g0), "s"(g1) : "memory");
    }

    // ---- stage z tile channels 32..63 with per-lane async global->LDS (ASYNCcnt path) ----
    for (int it = 0; it < 4; ++it) {
        int eidx = 4096 + it * 1024 + tid * 4;           // 4 floats / thread / iter
        int c = eidx >> 7, r = eidx & 127;
        unsigned lds_off = lds_base + (unsigned)((c * ZS_STRIDE + r) * 4);
        unsigned voff    = (unsigned)((c * 4096 + r) * 4);
        asm volatile("global_load_async_to_lds_b128 %0, %1, %2"
                     :: "v"(lds_off), "v"(voff), "s"(zb) : "memory");
    }

    // ---- overlap both DMAs: |e|^2 for all 1024 codes (emb is L2-resident, 256KB) ----
    for (int e = tid; e < NE; e += 256) {
        const float4* er = (const float4*)(emb + (size_t)e * CDIM);
        float s = 0.f;
#pragma unroll
        for (int q = 0; q < CDIM / 4; ++q) {
            float4 v = er[q];
            s += v.x * v.x + v.y * v.y + v.z * v.z + v.w * v.w;
        }
        esq_s[e] = s;
    }

    asm volatile("s_wait_asynccnt 0" ::: "memory");
    if (tid < 32) __builtin_amdgcn_s_wait_tensorcnt(0);
    __syncthreads();

    const int wave = tid >> 5;
    const int lane = tid & 31;
    const int m    = lane & 15;              // A: row ; B/C: column
    const int rowb = wave * 16;              // this wave's 16 rows
    const int kA0  = (lane < 16) ? 0 : 8;    // 16-bit A 16x32 per-lane K pattern

    // ---- persistent A fragments: K chunks {0..31, 32..63}, hi/lo bf16 split ----
    v16bf aH[2], aL[2];
#pragma unroll
    for (int kc = 0; kc < 2; ++kc) {
#pragma unroll
        for (int i = 0; i < 16; ++i) {
            int k = kc * 32 + 2 * kA0 + i;
            float x = zs[k * ZS_STRIDE + rowb + m];
            __bf16 h = (__bf16)x;
            aH[kc][i] = h;
            aL[kc][i] = (__bf16)(x - (float)h);
        }
    }

    float bestv[8];
    int   besti[8];
#pragma unroll
    for (int j = 0; j < 8; ++j) { bestv[j] = 3.4e38f; besti[j] = 0; }

    __bf16* bptr = (__bf16*)bfrag;

    for (int cb = 0; cb < NE; cb += ECHUNK) {
        __syncthreads();                     // previous chunk's tiles done before overwrite
        // ---- cooperative pack: B fragments in exact WMMA lane layout ----
        // word w = ((t*2+kc)*2+p)*256 + l*8 + j  -> bf16 elems 2w, 2w+1
        for (int w = tid; w < 4096; w += 256) {
            int j  = w & 7;
            int l  = (w >> 3) & 31;
            int p  = (w >> 8) & 1;           // 0 = hi, 1 = lo
            int kc = (w >> 9) & 1;
            int t  = w >> 10;
            int n  = l & 15;                 // column code within tile
            int kb = (l & 16);               // B 32x16: lanes 0-15 K 0..15, lanes 16-31 K 16..31
            int k  = kc * 32 + kb + 2 * j;
            const float* src = emb + (size_t)(cb + t * 16 + n) * CDIM + k;
            float x0 = src[0], x1 = src[1];
            __bf16 h0 = (__bf16)x0, h1 = (__bf16)x1;
            if (p) {
                h0 = (__bf16)(x0 - (float)h0);
                h1 = (__bf16)(x1 - (float)h1);
            }
            bptr[2 * w + 0] = h0;
            bptr[2 * w + 1] = h1;
        }
        if (cb + ECHUNK < NE)                // global_prefetch_b8 of next chunk
            __builtin_prefetch(&emb[(size_t)(cb + ECHUNK) * CDIM + tid * 16], 0, 1);
        __syncthreads();

#pragma unroll
        for (int t = 0; t < 4; ++t) {
            v8f acc = {};
#pragma unroll
            for (int kc = 0; kc < 2; ++kc) {
                v16bf bh = bfrag[((t * 2 + kc) * 2 + 0) * 32 + lane];  // 2x ds_load_b128
                v16bf bl = bfrag[((t * 2 + kc) * 2 + 1) * 32 + lane];
                acc = wmma_bf16(aH[kc], bh, acc);   // hi*hi
                acc = wmma_bf16(aH[kc], bl, acc);   // hi*lo
                acc = wmma_bf16(aL[kc], bh, acc);   // lo*hi
            }
            // score = |e|^2 - 2 z.e ; C-layout: one column code per lane
            int   code = cb + t * 16 + m;
            float eq   = esq_s[code];
#pragma unroll
            for (int j = 0; j < 8; ++j) {
                float s = eq - 2.0f * acc[j];
                if (s < bestv[j]) { bestv[j] = s; besti[j] = code; }
            }
        }
    }

    // ---- argmin across the 16 code-columns (lanes) of each half ----
#pragma unroll
    for (int j = 0; j < 8; ++j) {
        for (int d = 8; d > 0; d >>= 1) {
            float ov = __shfl_xor(bestv[j], d, 32);
            int   oi = __shfl_xor(besti[j], d, 32);
            if (ov < bestv[j] || (ov == bestv[j] && oi < besti[j])) { bestv[j] = ov; besti[j] = oi; }
        }
    }
    if (lane == 0) {
#pragma unroll
        for (int j = 0; j < 8; ++j) best_s[rowb + j] = besti[j];        // rows M=0..7
    }
    if (lane == 16) {
#pragma unroll
        for (int j = 0; j < 8; ++j) best_s[rowb + 8 + j] = besti[j];    // rows M=8..15
    }
    __syncthreads();

    // ---- epilogue: indices, z_q gather, loss partial, EMA segment sums ----
    float* out_zq  = out + OFF_ZQ;
    float* out_idx = out + OFF_IDX;
    float* esum    = ws + WS_ESUM;

    for (int r = tid; r < ROWS; r += 256) {
        out_idx[n0 + r] = (float)best_s[r];
        atomicAdd(&ws[WS_CNT + best_s[r]], 1.0f);
    }

    float lpart = 0.f;
    for (int idx = tid; idx < CDIM * ROWS; idx += 256) {
        int c = idx >> 7, r = idx & 127;
        int e = best_s[r];
        float q  = emb[(size_t)e * CDIM + c];    // L2-resident table
        float zv = zs[c * ZS_STRIDE + r];
        out_zq[b * 262144 + c * 4096 + o0 + r] = q;   // z_q_st == z_q numerically
        float dd = q - zv;
        lpart += dd * dd;
        atomicAdd(&esum[e * CDIM + c], zv);
    }
    for (int d = 16; d > 0; d >>= 1) lpart += __shfl_xor(lpart, d, 32);
    if (lane == 0) wred[wave] = lpart;
    __syncthreads();
    if (tid == 0) {
        float s = 0.f;
        for (int w = 0; w < 8; ++w) s += wred[w];
        atomicAdd(&ws[WS_SSE], s);
    }
}

__global__ void init_ws_kernel(float* __restrict__ ws) {
    int i = blockIdx.x * blockDim.x + threadIdx.x;
    if (i < WS_ZERON) ws[i] = 0.f;
}

// new_cluster_size + global n = sum(new_cluster_size)
__global__ void ema_cs_kernel(const float* __restrict__ cs,
                              float* __restrict__ out, float* __restrict__ ws) {
    int e = blockIdx.x * blockDim.x + threadIdx.x;
    float ncs = 0.f;
    if (e < NE) {
        ncs = 0.99f * cs[e] + 0.01f * ws[WS_CNT + e];
        out[OFF_NCS + e] = ncs;
    }
    for (int d = 16; d > 0; d >>= 1) ncs += __shfl_xor(ncs, d, 32);
    if ((threadIdx.x & 31) == 0) atomicAdd(&ws[WS_N], ncs);
}

// new_embed_avg, normalized new_embedding, final loss scale
__global__ void ema_final_kernel(const float* __restrict__ ea,
                                 float* __restrict__ out, const float* __restrict__ ws) {
    int i = blockIdx.x * blockDim.x + threadIdx.x;
    if (i < NE * CDIM) {
        int e = i >> 6;
        float nea = 0.99f * ea[i] + 0.01f * ws[WS_ESUM + i];
        out[OFF_NEA + i] = nea;
        float n   = ws[WS_N];
        float ncs = out[OFF_NCS + e];
        float csn = (ncs + 1e-5f) / (n + NE * 1e-5f) * n;
        out[OFF_NEMB + i] = nea / csn;
    }
    if (i == 0) out[OFF_LOSS] = 0.25f * ws[WS_SSE] / 4194304.0f;
}

extern "C" void kernel_launch(void* const* d_in, const int* in_sizes, int n_in,
                              void* d_out, int out_size, void* d_ws, size_t ws_size,
                              hipStream_t stream) {
    const float* z   = (const float*)d_in[0];   // (16,64,64,64)
    const float* emb = (const float*)d_in[1];   // (1024,64)
    const float* cs  = (const float*)d_in[2];   // (1024,)
    const float* ea  = (const float*)d_in[3];   // (1024,64)
    float* out = (float*)d_out;
    float* ws  = (float*)d_ws;

    init_ws_kernel<<<(WS_ZERON + 255) / 256, 256, 0, stream>>>(ws);
    vq_wmma_argmin_kernel<<<NPIX / ROWS, 256, 0, stream>>>(z, emb, out, ws);
    ema_cs_kernel<<<NE / 256, 256, 0, stream>>>(cs, out, ws);
    ema_final_kernel<<<(NE * CDIM) / 256, 256, 0, stream>>>(ea, out, ws);
}
